// NonlinearFilterWithInput_82051055223039
// MI455X (gfx1250) — matrix operand order repaired and checked by reading, not compiled, not run on hardware
//
#include <hip/hip_runtime.h>

#define BN 64
#define TN 256
#define SN 16
#define LN 128
#define RN 16
#define HN 256
#define UEN 64

typedef __attribute__((ext_vector_type(16))) __bf16 v16bf;
typedef __attribute__((ext_vector_type(8)))  float  v8f;
typedef __attribute__((ext_vector_type(4)))  unsigned int v4u;

union BFrag { v16bf v; v4u q[2]; unsigned short u[16]; };

__device__ __forceinline__ unsigned short f2bf(float x) {
  unsigned int b = __float_as_uint(x);
  return (unsigned short)((b + 0x7FFFu + ((b >> 16) & 1u)) >> 16);  // RNE
}

// CDNA5 VALU has a native TANH transcendental; prefer it, else clamped
// rational form with native exp (v_exp_f32) -- both avoid libm's branchy path.
__device__ __forceinline__ float fast_tanh(float x) {
#if __has_builtin(__builtin_amdgcn_tanhf)
  return __builtin_amdgcn_tanhf(x);
#else
  float xc = fminf(fmaxf(x, -9.0f), 9.0f);
  float e  = __expf(2.0f * xc);
  return (e - 1.0f) / (e + 1.0f);
#endif
}

// Pre-swizzle W1/W2 (f32 -> bf16) into WMMA B-fragment order in d_ws so every
// B operand is a single coalesced 32B load per lane (stays hot in 192MB L2).
// Layout per ISA 16-bit B 32x16: lane n = lane&15, k = e<8 ? (lane>>4)*8 + e : 16 + (lane>>4)*8 + (e-8).
__global__ __launch_bounds__(256) void swizzle_weights(
    const float* __restrict__ W1, const float* __restrict__ W2,
    unsigned short* __restrict__ wf) {
  int idx = blockIdx.x * 256 + threadIdx.x;      // 0..65535
  if (idx < 32768) {                             // W1: 16 ntiles x 4 kchunks
    int e = idx & 15, lane = (idx >> 4) & 31, kc = (idx >> 9) & 3, nt = idx >> 11;
    int kk = kc * 32 + ((e < 8) ? ((lane >> 4) * 8 + e) : (16 + (lane >> 4) * 8 + (e - 8)));
    int n  = nt * 16 + (lane & 15);
    wf[idx] = f2bf(W1[kk * HN + n]);
  } else {                                       // W2: 8 ntiles x 8 kchunks
    int j = idx - 32768;
    int e = j & 15, lane = (j >> 4) & 31, kc = (j >> 9) & 7, nt = j >> 12;
    int kk = kc * 32 + ((e < 8) ? ((lane >> 4) * 8 + e) : (16 + (lane >> 4) * 8 + (e - 8)));
    int n  = nt * 16 + (lane & 15);
    wf[idx] = f2bf(W2[kk * LN + n]);
  }
}

__global__ __launch_bounds__(256) void nlf_kernel(
    const float* __restrict__ u_g, const float* __restrict__ k_g,
    const float* __restrict__ K_g, const float* __restrict__ ez_g,
    const float* __restrict__ ew_g, const float* __restrict__ minit_g,
    const float* __restrict__ Qd_g, const float* __restrict__ Q0_g,
    const unsigned short* __restrict__ wf, float* __restrict__ out) {
  const int b    = blockIdx.x;          // one workgroup per batch (independent scans)
  const int tid  = threadIdx.x;
  const int lane = tid & 31;
  const int wv   = tid >> 5;            // 8 wave32 per workgroup
  const int half = lane >> 4;
  const int l15  = lane & 15;

  __shared__ float zfS[SN * LN];                              // z_prev f32
  __shared__ __align__(16) unsigned short zbS[SN * LN];       // z_prev bf16 (A-frag feed)
  __shared__ __align__(16) unsigned short hbS[SN * HN];       // tanh(zW1) bf16
  __shared__ float mbS[SN * LN];                              // m
  __shared__ __align__(16) unsigned short KtTbS[8 * 32 * 16]; // K^T (K=32 zero-pad) B-frag, 8 ntiles
  __shared__ __align__(16) unsigned short KtbS[4 * 32 * 16];  // K_t in B-frag order
  __shared__ __align__(16) unsigned short KqTS[4 * 32 * 16];  // (qa*K_t)^T in A-frag order
  __shared__ float ktS[LN];
  __shared__ float utS[UEN];
  __shared__ float zpS[SN * LN];
  __shared__ float ewS[SN * RN];
  __shared__ float AinvS[RN * RN];
  __shared__ float w1S[SN * RN], w2S[SN * RN];
  __shared__ __align__(16) float v1S[SN * RN];
  __shared__ __align__(16) float v2S[SN * RN];
  __shared__ float qdS[LN], q0S[LN], sqS[LN], sq0S[LN];
  __shared__ float mfS[LN];
  __shared__ float qpS[SN];
  __shared__ float scal[4];   // [0]=logdet, [1]=trace accum, [2]=qp accum

  if (tid < LN) {
    float q = Qd_g[tid];  qdS[tid] = q;  sqS[tid]  = sqrtf(q);
    float q0 = Q0_g[tid]; q0S[tid] = q0; sq0S[tid] = sqrtf(q0);
  }
  __syncthreads();

  const size_t ZF_OFF = 0;
  const size_t MF_OFF = (size_t)SN * BN * TN * LN;
  const size_t ZP_OFF = MF_OFF + (size_t)BN * TN * LN;
  const size_t KL_OFF = ZP_OFF + (size_t)SN * BN * TN * LN;

  #pragma clang loop unroll(disable)
  for (int t = 0; t < TN; ++t) {
    const bool t0 = (t == 0);
    const size_t kbase = ((size_t)b * TN + t) * (LN * RN);

    // ---------- Phase L: stage step inputs into LDS ----------
    if (tid < LN)  ktS[tid] = k_g[((size_t)b * TN + t) * LN + tid];
    if (tid < UEN) utS[tid] = u_g[((size_t)b * TN + t) * UEN + tid];
    { int s = tid >> 4, r = tid & 15;
      ewS[tid] = ew_g[(((size_t)t * SN + s) * BN + b) * RN + r]; }
    if (tid < SN) qpS[tid] = 0.0f;
    if (tid < LN) mfS[tid] = 0.0f;
    if (tid < 2)  scal[1 + tid] = 0.0f;
    #pragma unroll
    for (int c = 0; c < 8; ++c) {
      int f = tid + 256 * c;
      { // z_p = sqrt(qa) * eps_z ; stream z_p straight to global output
        int s = f >> 7, l = f & 127;
        float e = ez_g[(((size_t)t * SN + s) * BN + b) * LN + l];
        float v = (t0 ? sq0S[l] : sqS[l]) * e;
        zpS[f] = v;
        out[ZP_OFF + (((size_t)(s * BN + b)) * TN + t) * LN + l] = v;
      }
      { // K^T in B-frag order (k = r in 0..15, zero-padded to K=32)
        #pragma unroll
        for (int hh = 0; hh < 2; ++hh) {
          int idx = f + hh * 2048;
          int e = idx & 15, ln = (idx >> 4) & 31, nt = (idx >> 9) & 7;
          unsigned short vb = 0;
          if (e < 8) vb = f2bf(K_g[kbase + (size_t)(nt * 16 + (ln & 15)) * RN + (ln >> 4) * 8 + e]);
          KtTbS[idx] = vb;
        }
      }
      { // bf16 fragment tables for K_t
        int e = f & 15, ln = (f >> 4) & 31, kc = f >> 9;
        int hv = ln >> 4;
        int lB = kc * 32 + ((e < 8) ? (hv * 8 + e) : (16 + hv * 8 + (e - 8))); // B-frag k index
        KtbS[f] = f2bf(K_g[kbase + (size_t)lB * RN + (ln & 15)]);
        int eA = (e < 8) ? (hv * 8 + e) : (16 + hv * 8 + (e - 8));
        int lA = kc * 32 + eA;                                // A-frag k index
        float qa = t0 ? q0S[lA] : qdS[lA];
        KqTS[f] = f2bf(qa * K_g[kbase + (size_t)lA * RN + (ln & 15)]);
      }
    }
    if (t + 1 < TN) {  // global_prefetch_b8 for next step's tiles
      __builtin_prefetch(K_g + kbase + LN * RN + tid * 8, 0, 1);
      __builtin_prefetch(ez_g + (((size_t)(t + 1) * SN + (tid >> 4)) * BN + b) * LN + (tid & 15) * 8, 0, 1);
    }
    __syncthreads();

    // ---------- Phase M: m = mean_fn(z_prev) + u  (bf16 WMMA, f32 accum) ----------
    if (t0) {
      #pragma unroll
      for (int c = 0; c < 8; ++c) {
        int f = tid + 256 * c, l = f & 127;
        mbS[f] = minit_g[l] + (l < UEN ? utS[l] : 0.0f);   // replicated across samples
      }
    } else {
      { // GEMM1: h = tanh(z @ W1); each wave owns 2 n-tiles of 16
        v8f a0 = {}, a1 = {};
        #pragma unroll
        for (int kc = 0; kc < 4; ++kc) {
          BFrag af;
          int base = l15 * LN + kc * 32 + half * 8;
          af.q[0] = *(const v4u*)&zbS[base];
          af.q[1] = *(const v4u*)&zbS[base + 16];
          BFrag b0, b1;
          const unsigned short* p0 = wf + (((2 * wv + 0) * 4 + kc) * 32 + lane) * 16;
          const unsigned short* p1 = wf + (((2 * wv + 1) * 4 + kc) * 32 + lane) * 16;
          b0.q[0] = *(const v4u*)p0; b0.q[1] = *(const v4u*)(p0 + 8);
          b1.q[0] = *(const v4u*)p1; b1.q[1] = *(const v4u*)(p1 + 8);
          a0 = __builtin_amdgcn_wmma_f32_16x16x32_bf16(false, af.v, false, b0.v, (short)0, a0, false, false);
          a1 = __builtin_amdgcn_wmma_f32_16x16x32_bf16(false, af.v, false, b1.v, (short)0, a1, false, false);
        }
        #pragma unroll
        for (int i = 0; i < 8; ++i) {
          int row = i + half * 8;
          hbS[row * HN + (2 * wv + 0) * 16 + l15] = f2bf(fast_tanh(a0[i]));
          hbS[row * HN + (2 * wv + 1) * 16 + l15] = f2bf(fast_tanh(a1[i]));
        }
      }
      __syncthreads();
      { // GEMM2: m = z + h @ W2 (+ u on first 64 cols, preloaded into C)
        int coln = wv * 16 + l15;
        v8f acc = {};
        #pragma unroll
        for (int i = 0; i < 8; ++i) {
          int row = i + half * 8;
          acc[i] = zfS[row * LN + coln] + (coln < UEN ? utS[coln] : 0.0f);
        }
        #pragma unroll
        for (int kc = 0; kc < 8; ++kc) {
          BFrag af;
          int base = l15 * HN + kc * 32 + half * 8;
          af.q[0] = *(const v4u*)&hbS[base];
          af.q[1] = *(const v4u*)&hbS[base + 16];
          BFrag bw;
          const unsigned short* p = wf + 32768 + ((wv * 8 + kc) * 32 + lane) * 16;
          bw.q[0] = *(const v4u*)p; bw.q[1] = *(const v4u*)(p + 8);
          acc = __builtin_amdgcn_wmma_f32_16x16x32_bf16(false, af.v, false, bw.v, (short)0, acc, false, false);
        }
        #pragma unroll
        for (int i = 0; i < 8; ++i)
          mbS[(i + half * 8) * LN + coln] = acc[i];
      }
    }
    __syncthreads();

    // ---------- Phase A/G: concurrent per-wave work ----------
    if (wv == 0) {
      // A = I + Kq^T K via WMMA (C preloaded with identity)
      v8f acc;
      #pragma unroll
      for (int i = 0; i < 8; ++i) acc[i] = ((i + half * 8) == l15) ? 1.0f : 0.0f;
      #pragma unroll
      for (int kc = 0; kc < 4; ++kc) {
        BFrag af, bk;
        const unsigned short* pa = &KqTS[(kc * 32 + lane) * 16];
        const unsigned short* pb = &KtbS[(kc * 32 + lane) * 16];
        af.q[0] = *(const v4u*)pa; af.q[1] = *(const v4u*)(pa + 8);
        bk.q[0] = *(const v4u*)pb; bk.q[1] = *(const v4u*)(pb + 8);
        acc = __builtin_amdgcn_wmma_f32_16x16x32_bf16(false, af.v, false, bk.v, (short)0, acc, false, false);
      }
      // Redistribute D fragment into per-lane columns of [A | I]
      float col[16];
      #pragma unroll
      for (int i = 0; i < 8; ++i) {
        float x = acc[i];
        col[i]     = __shfl(x, l15, 32);
        col[i + 8] = __shfl(x, l15 + 16, 32);
      }
      if (lane >= 16) {
        #pragma unroll
        for (int i = 0; i < 16; ++i) col[i] = (i == lane - 16) ? 1.0f : 0.0f;
      }
      // Shuffle-based Gauss-Jordan (SPD, no pivoting); pivots -> logdet.
      float logdet = 0.0f;
      #pragma unroll
      for (int p = 0; p < 16; ++p) {
        float piv = __shfl(col[p], p, 32);
        logdet += logf(piv);
        float pinv = 1.0f / piv;
        col[p] *= pinv;
        #pragma unroll
        for (int i = 0; i < 16; ++i) {
          if (i == p) continue;
          float fi = __shfl(col[i], p, 32);
          col[i] -= fi * col[p];
        }
      }
      if (lane >= 16) {
        #pragma unroll
        for (int i = 0; i < 16; ++i) AinvS[i * 16 + (lane - 16)] = col[i];
      }
      if (lane == 0) scal[0] = logdet;
    } else if (wv == 1) {
      // w1 = K^T (m + Q_diag*k)  [reference uses Q_diag here even at t=0]
      v8f acc = {};
      #pragma unroll
      for (int kc = 0; kc < 4; ++kc) {
        BFrag af;
        #pragma unroll
        for (int e = 0; e < 16; ++e) {
          int l = kc * 32 + ((e < 8) ? (half * 8 + e) : (16 + half * 8 + e - 8));
          af.u[e] = f2bf(mbS[l15 * LN + l] + qdS[l] * ktS[l]);
        }
        BFrag bk;
        const unsigned short* pb = &KtbS[(kc * 32 + lane) * 16];
        bk.q[0] = *(const v4u*)pb; bk.q[1] = *(const v4u*)(pb + 8);
        acc = __builtin_amdgcn_wmma_f32_16x16x32_bf16(false, af.v, false, bk.v, (short)0, acc, false, false);
      }
      #pragma unroll
      for (int i = 0; i < 8; ++i) w1S[(i + half * 8) * RN + l15] = acc[i];
    } else if (wv == 2) {
      // w2 = K^T z_p + eps_w  (eps_w preloaded into C)
      v8f acc;
      #pragma unroll
      for (int i = 0; i < 8; ++i) acc[i] = ewS[(i + half * 8) * RN + l15];
      #pragma unroll
      for (int kc = 0; kc < 4; ++kc) {
        BFrag af;
        #pragma unroll
        for (int e = 0; e < 16; ++e) {
          int l = kc * 32 + ((e < 8) ? (half * 8 + e) : (16 + half * 8 + e - 8));
          af.u[e] = f2bf(zpS[l15 * LN + l]);
        }
        BFrag bk;
        const unsigned short* pb = &KtbS[(kc * 32 + lane) * 16];
        bk.q[0] = *(const v4u*)pb; bk.q[1] = *(const v4u*)(pb + 8);
        acc = __builtin_amdgcn_wmma_f32_16x16x32_bf16(false, af.v, false, bk.v, (short)0, acc, false, false);
      }
      #pragma unroll
      for (int i = 0; i < 8; ++i) w2S[(i + half * 8) * RN + l15] = acc[i];
    }
    __syncthreads();

    // ---------- Phase C: v = A^{-1} w (one (s,r) per thread) ----------
    {
      int s = tid >> 4, r = tid & 15;
      float a1 = 0.0f, a2 = 0.0f;
      #pragma unroll
      for (int q = 0; q < 16; ++q) {
        float ai = AinvS[r * 16 + q];
        a1 = fmaf(ai, w1S[s * 16 + q], a1);
        a2 = fmaf(ai, w2S[s * 16 + q], a2);
      }
      v1S[tid] = a1; v2S[tid] = a2;
    }
    __syncthreads();

    // ---------- Phase F: Kv1/Kv2 as WMMA (K=16 zero-padded), then epilogue ----------
    {
      // B fragment: this wave's 16 columns of K^T (shared by both WMMAs)
      BFrag bk;
      const unsigned short* pb = &KtTbS[(wv * 32 + lane) * 16];
      bk.q[0] = *(const v4u*)pb; bk.q[1] = *(const v4u*)(pb + 8);
      // A fragments: v1/v2, rows = samples, k = r (upper 16 k zero)
      BFrag a1, a2;
      {
        const float4* pv1 = (const float4*)&v1S[l15 * 16 + half * 8];
        const float4* pv2 = (const float4*)&v2S[l15 * 16 + half * 8];
        float4 x0 = pv1[0], x1 = pv1[1];
        float4 y0 = pv2[0], y1 = pv2[1];
        a1.u[0] = f2bf(x0.x); a1.u[1] = f2bf(x0.y); a1.u[2] = f2bf(x0.z); a1.u[3] = f2bf(x0.w);
        a1.u[4] = f2bf(x1.x); a1.u[5] = f2bf(x1.y); a1.u[6] = f2bf(x1.z); a1.u[7] = f2bf(x1.w);
        a2.u[0] = f2bf(y0.x); a2.u[1] = f2bf(y0.y); a2.u[2] = f2bf(y0.z); a2.u[3] = f2bf(y0.w);
        a2.u[4] = f2bf(y1.x); a2.u[5] = f2bf(y1.y); a2.u[6] = f2bf(y1.z); a2.u[7] = f2bf(y1.w);
        #pragma unroll
        for (int e = 8; e < 16; ++e) { a1.u[e] = 0; a2.u[e] = 0; }
      }
      v8f z8 = {};
      v8f d1 = __builtin_amdgcn_wmma_f32_16x16x32_bf16(false, a1.v, false, bk.v, (short)0, z8, false, false);
      v8f d2 = __builtin_amdgcn_wmma_f32_16x16x32_bf16(false, a2.v, false, bk.v, (short)0, z8, false, false);
      const int l = wv * 16 + l15;                 // this lane's column
      float qa = t0 ? q0S[l] : qdS[l];
      float kt = ktS[l];
      float msum = 0.0f;
      #pragma unroll
      for (int i = 0; i < 8; ++i) {
        int s = i + half * 8;                      // this lane-half's sample rows
        float g  = kt - d1[i];
        float mf = mbS[s * LN + l] + qa * g;
        float zv = mf + zpS[s * LN + l] - d2[i];
        zfS[s * LN + l] = zv;
        zbS[s * LN + l] = f2bf(zv);
        out[ZF_OFF + (((size_t)(s * BN + b)) * TN + t) * LN + l] = zv;
        msum += mf;
        float qv = qa * g * g;                     // reduce over l within half-wave
        qv += __shfl_xor(qv, 1, 32);
        qv += __shfl_xor(qv, 2, 32);
        qv += __shfl_xor(qv, 4, 32);
        qv += __shfl_xor(qv, 8, 32);
        if (l15 == 0) atomicAdd(&qpS[s], qv);
      }
      atomicAdd(&mfS[l], msum * 0.0625f);          // mean over S=16 samples
    }
    __syncthreads();

    // ---------- Phase K: kl (0.5*(mean qp + trace(Ainv)-R + logdet)), m_f mean ----------
    if (tid < 16) {
      atomicAdd(&scal[1], AinvS[tid * 16 + tid]);
      atomicAdd(&scal[2], qpS[tid]);
    }
    __syncthreads();
    if (tid == 0) {
      float tr = scal[1] - (float)RN;
      out[KL_OFF + (size_t)b * TN + t] = 0.5f * (scal[2] * 0.0625f + tr + scal[0]);
    }
    if (tid < LN)
      out[MF_OFF + (((size_t)b * TN) + t) * LN + tid] = mfS[tid];
    __syncthreads();
  }
}

extern "C" void kernel_launch(void* const* d_in, const int* in_sizes, int n_in,
                              void* d_out, int out_size, void* d_ws, size_t ws_size,
                              hipStream_t stream) {
  const float* u      = (const float*)d_in[0];
  const float* k      = (const float*)d_in[1];
  const float* K      = (const float*)d_in[2];
  const float* eps_z  = (const float*)d_in[3];
  const float* eps_w  = (const float*)d_in[4];
  const float* W1     = (const float*)d_in[5];
  const float* W2     = (const float*)d_in[6];
  const float* m_init = (const float*)d_in[7];
  const float* Qd     = (const float*)d_in[8];
  const float* Q0     = (const float*)d_in[9];
  unsigned short* wf  = (unsigned short*)d_ws;   // 128 KB of bf16 fragment tables
  float* out          = (float*)d_out;

  swizzle_weights<<<256, 256, 0, stream>>>(W1, W2, wf);
  nlf_kernel<<<BN, 256, 0, stream>>>(u, k, K, eps_z, eps_w, m_init, Qd, Q0, wf, out);
}
